// CrossMultiAttentionHead_9371618640058
// MI455X (gfx1250) — compile-verified, matrix operand order checked by reading, not run on hardware
//
#include <hip/hip_runtime.h>
#include <hip/hip_bf16.h>
#include <stdint.h>

// ---------------------------------------------------------------------------
// CrossMultiAttentionHead for MI455X (gfx1250, wave32, WMMA bf16 16x16x32)
//   B=4, T=2048, C=2048, 16 heads x 128 head-dim, causal.
// Pipeline:
//   1) f32 -> bf16 convert activations; transpose-convert weights to [N][K]
//   2) WMMA GEMM: KV = Xenc*Wkv+b (bf16 out), Q = Xdec*Wq+b (bf16 out)
//      (tiles staged with GLOBAL_LOAD_ASYNC_TO_LDS_B128, ASYNCcnt-tracked)
//   3) Flash attention (WMMA QK^T + online softmax + WMMA PV) -> bf16
//   4) WMMA GEMM: out = AttnOut*Wo + b_o (f32 out)
// ---------------------------------------------------------------------------

typedef __bf16 bf16_t;
typedef __attribute__((ext_vector_type(16))) __bf16 v16bf;
typedef __attribute__((ext_vector_type(8)))  __bf16 v8bf;
typedef __attribute__((ext_vector_type(8)))  float  v8f;

#define DIMC   2048
#define NHEADS 16
#define HEADD  128
#define BATCH  4
#define SEQ    2048
#define MTOT   (BATCH * SEQ)   // 8192 rows total

// round-to-nearest-even f32 -> bf16 (pure integer path)
static __device__ __forceinline__ bf16_t f2bf(float x) {
  union { float f; unsigned u; } a; a.f = x;
  unsigned r = a.u + 0x7FFFu + ((a.u >> 16) & 1u);
  union { unsigned short s; bf16_t b; } o;
  o.s = (unsigned short)(r >> 16);
  return o.b;
}

// gfx1250 async DMA: 16 bytes global -> LDS, tracked by ASYNCcnt.
// Inline asm (not the clang builtin) so it is portable across toolchains.
static __device__ __forceinline__ void async_copy_b128(const bf16_t* gsrc,
                                                       bf16_t* ldst) {
  unsigned lds = (unsigned)(uintptr_t)ldst;
  asm volatile("global_load_async_to_lds_b128 %0, %1, off"
               :: "v"(lds), "v"(gsrc)
               : "memory");
}
static __device__ __forceinline__ void wait_async0() {
  asm volatile("s_wait_asynccnt 0x0" ::: "memory");
}

// Build a 16x32-layout WMMA fragment for lane `l`:
//   vector elems 0..7  = K chunk at koff      (contiguous)
//   vector elems 8..15 = K chunk at koff + 16 (contiguous)
static __device__ __forceinline__ v16bf load_frag(const bf16_t* p) {
  union { v16bf v16; v8bf v8[2]; } u;
  u.v8[0] = *reinterpret_cast<const v8bf*>(p);
  u.v8[1] = *reinterpret_cast<const v8bf*>(p + 16);
  return u.v16;
}

// ---------------------------------------------------------------------------
// elementwise f32 -> bf16
// ---------------------------------------------------------------------------
__global__ __launch_bounds__(256) void cvt_bf16_kernel(
    const float* __restrict__ x, bf16_t* __restrict__ y, int n) {
  int i = blockIdx.x * 256 + threadIdx.x;
  if (i < n) y[i] = f2bf(x[i]);
}

// Wt[n*K + k] = (bf16) W[k*N + n]   (weights stored transposed, K contiguous)
__global__ __launch_bounds__(256) void cvt_transpose_kernel(
    const float* __restrict__ w, bf16_t* __restrict__ wt, int K, int N) {
  int i = blockIdx.x * 256 + threadIdx.x;
  if (i < K * N) {
    int n = i / K;
    int k = i - n * K;
    wt[i] = f2bf(w[(size_t)k * N + n]);
  }
}

// ---------------------------------------------------------------------------
// GEMM: C[M][N] = A[M][K] * Bt[N][K]^T + bias[N]
//   128x128 block tile, 256 threads (8 waves), wave tile 32x64 (2x4 WMMA).
//   Output either f32 (outF, final proj) or bf16 (outB, K/V/Q paths).
// ---------------------------------------------------------------------------
__global__ __launch_bounds__(256) void gemm_bf16_wmma_kernel(
    const bf16_t* __restrict__ A, const bf16_t* __restrict__ Bt,
    const float* __restrict__ bias, float* __restrict__ outF,
    bf16_t* __restrict__ outB, int M, int N, int K, int ldc) {
  __shared__ bf16_t As[128 * 32];   // 8 KB
  __shared__ bf16_t Bs[128 * 32];   // 8 KB

  const int tid   = threadIdx.x;
  const int lane  = tid & 31;
  const int wid   = tid >> 5;
  const int bM    = blockIdx.y * 128;
  const int bN    = blockIdx.x * 128;
  const int wMoff = (wid & 3) * 32;   // 2 m-tiles of 16
  const int wNoff = (wid >> 2) * 64;  // 4 n-tiles of 16
  const int lrid  = lane & 15;
  const int khalf = (lane >> 4) * 8;

  // staging chunk assignment (fixed per thread): 2 chunks per matrix
  const int c0row = tid >> 2, c0ko = (tid & 3) * 8;
  const int c1row = (tid + 256) >> 2, c1ko = ((tid + 256) & 3) * 8;

  v8f acc[2][4];
#pragma unroll
  for (int mt = 0; mt < 2; ++mt)
#pragma unroll
    for (int nt = 0; nt < 4; ++nt) acc[mt][nt] = {};

  for (int k0 = 0; k0 < K; k0 += 32) {
    // ---- async DMA stage: memory -> LDS without touching VGPRs
    async_copy_b128(&A[(size_t)(bM + c0row) * K + k0 + c0ko], &As[c0row * 32 + c0ko]);
    async_copy_b128(&A[(size_t)(bM + c1row) * K + k0 + c1ko], &As[c1row * 32 + c1ko]);
    async_copy_b128(&Bt[(size_t)(bN + c0row) * K + k0 + c0ko], &Bs[c0row * 32 + c0ko]);
    async_copy_b128(&Bt[(size_t)(bN + c1row) * K + k0 + c1ko], &Bs[c1row * 32 + c1ko]);
    if (k0 + 32 < K) {  // pull next tiles toward the near caches
      __builtin_prefetch(&A[(size_t)(bM + c0row) * K + k0 + 32], 0, 3);
      __builtin_prefetch(&Bt[(size_t)(bN + c0row) * K + k0 + 32], 0, 3);
    }
    wait_async0();
    __syncthreads();

    v16bf af[2], bfr[4];
#pragma unroll
    for (int mt = 0; mt < 2; ++mt)
      af[mt] = load_frag(&As[(wMoff + mt * 16 + lrid) * 32 + khalf]);
#pragma unroll
    for (int nt = 0; nt < 4; ++nt)
      bfr[nt] = load_frag(&Bs[(wNoff + nt * 16 + lrid) * 32 + khalf]);

#pragma unroll
    for (int mt = 0; mt < 2; ++mt)
#pragma unroll
      for (int nt = 0; nt < 4; ++nt)
        acc[mt][nt] = __builtin_amdgcn_wmma_f32_16x16x32_bf16(
            false, af[mt], false, bfr[nt], (short)0, acc[mt][nt], false, false);
    __syncthreads();
  }

  // ---- epilogue: lane holds col n = lrid, rows m = r + 8*(lane/16)
  const int mrb = (lane >> 4) * 8;
#pragma unroll
  for (int mt = 0; mt < 2; ++mt) {
#pragma unroll
    for (int nt = 0; nt < 4; ++nt) {
      int n    = bN + wNoff + nt * 16 + lrid;
      float bv = bias[n];
#pragma unroll
      for (int r = 0; r < 8; ++r) {
        int m   = bM + wMoff + mt * 16 + mrb + r;
        float v = acc[mt][nt][r] + bv;
        if (outF) outF[(size_t)m * ldc + n] = v;
        else      outB[(size_t)m * ldc + n] = f2bf(v);
      }
    }
  }
}

// ---------------------------------------------------------------------------
// Flash attention (causal): one workgroup = 64 query rows of one (b, h).
//   4 waves, wave w owns q rows w*16..w*16+15. Key blocks of 64 streamed.
//   KV layout: [B*T][2*DIMC] (cols 0..2047 = K, 2048..4095 = V)
// ---------------------------------------------------------------------------
__global__ __launch_bounds__(128) void flash_attn_wmma_kernel(
    const bf16_t* __restrict__ KV, const bf16_t* __restrict__ Q,
    bf16_t* __restrict__ O) {
  __shared__ bf16_t Qs[64 * 128];       // 16 KB
  __shared__ bf16_t Ks[64 * 128];       // 16 KB
  __shared__ bf16_t Vts[128 * 64];      // 16 KB (transposed: [d][key])
  __shared__ bf16_t Ps[4 * 16 * 64];    //  8 KB (per-wave P tile)

  const int tid   = threadIdx.x;
  const int lane  = tid & 31;
  const int w     = tid >> 5;          // wave 0..3
  const int qb    = blockIdx.x;        // query block (64 rows)
  const int h     = blockIdx.y;
  const int b     = blockIdx.z;
  const int lrid  = lane & 15;
  const int lhalf = lane >> 4;
  const int khalf = lhalf * 8;
  const float scale = 0.08838834764831845f;  // 1/sqrt(128)

  const size_t qrow0 = (size_t)b * SEQ + (size_t)qb * 64;
  const bf16_t* Qg = Q + qrow0 * DIMC + h * HEADD;
  const bf16_t* Kg = KV + (size_t)b * SEQ * (2 * DIMC) + h * HEADD;
  const bf16_t* Vg = Kg + DIMC;

  // stage Q tile once via async DMA: 64x128, 1024 chunks of 8, 8 per thread
#pragma unroll
  for (int i = 0; i < 8; ++i) {
    int c   = tid + 128 * i;
    int row = c >> 4;
    int co  = (c & 15) * 8;
    async_copy_b128(&Qg[(size_t)row * DIMC + co], &Qs[row * 128 + co]);
  }

  float mrow[8], lrow[8];
  v8f oacc[8];
#pragma unroll
  for (int r = 0; r < 8; ++r) { mrow[r] = -3.0e38f; lrow[r] = 0.f; }
#pragma unroll
  for (int nt = 0; nt < 8; ++nt) oacc[nt] = {};

  for (int kb = 0; kb <= qb; ++kb) {
    __syncthreads();  // previous-iteration LDS consumers done before restage
    const bf16_t* Kt = Kg + (size_t)(kb * 64) * (2 * DIMC);
    const bf16_t* Vt = Vg + (size_t)(kb * 64) * (2 * DIMC);
#pragma unroll
    for (int i = 0; i < 8; ++i) {
      int c   = tid + 128 * i;
      int row = c >> 4;
      int co  = (c & 15) * 8;
      // K tile: straight copy -> async DMA path
      async_copy_b128(&Kt[(size_t)row * (2 * DIMC) + co], &Ks[row * 128 + co]);
      // V tile: transpose during staging (per-element scatter)
      union { v8bf v; bf16_t e[8]; } uv;
      uv.v = *reinterpret_cast<const v8bf*>(&Vt[(size_t)row * (2 * DIMC) + co]);
#pragma unroll
      for (int e = 0; e < 8; ++e) Vts[(co + e) * 64 + row] = uv.e[e];
    }
    wait_async0();   // Q (first iter) + K DMA complete
    __syncthreads();

    // ---- S = Q K^T : 4 n-tiles (64 keys) x 4 contraction steps over d=128
    v8f s[4];
#pragma unroll
    for (int nt = 0; nt < 4; ++nt) s[nt] = {};
#pragma unroll
    for (int kc = 0; kc < 4; ++kc) {
      v16bf aq = load_frag(&Qs[(w * 16 + lrid) * 128 + kc * 32 + khalf]);
#pragma unroll
      for (int nt = 0; nt < 4; ++nt) {
        v16bf bk = load_frag(&Ks[(nt * 16 + lrid) * 128 + kc * 32 + khalf]);
        s[nt] = __builtin_amdgcn_wmma_f32_16x16x32_bf16(
            false, aq, false, bk, (short)0, s[nt], false, false);
      }
    }

    // ---- scale + causal mask + row max (rows r live in lane-half lhalf)
    const int qg0 = qb * 64 + w * 16 + lhalf * 8;
    const int kb0 = kb * 64;
    float rmax[8];
#pragma unroll
    for (int r = 0; r < 8; ++r) rmax[r] = -3.0e38f;
#pragma unroll
    for (int nt = 0; nt < 4; ++nt) {
      int key = kb0 + nt * 16 + lrid;
#pragma unroll
      for (int r = 0; r < 8; ++r) {
        float v = s[nt][r] * scale;
        if (key > qg0 + r) v = -3.0e38f;
        s[nt][r] = v;
        rmax[r] = fmaxf(rmax[r], v);
      }
    }
#pragma unroll
    for (int off = 1; off < 16; off <<= 1)
#pragma unroll
      for (int r = 0; r < 8; ++r)
        rmax[r] = fmaxf(rmax[r], __shfl_xor(rmax[r], off, 32));

    // ---- online softmax update
    float corr[8], rsum[8];
#pragma unroll
    for (int r = 0; r < 8; ++r) {
      float mnew = fmaxf(mrow[r], rmax[r]);
      corr[r] = __expf(mrow[r] - mnew);
      mrow[r] = mnew;
      rsum[r] = 0.f;
    }
#pragma unroll
    for (int nt = 0; nt < 4; ++nt)
#pragma unroll
      for (int r = 0; r < 8; ++r) {
        float p = __expf(s[nt][r] - mrow[r]);
        s[nt][r] = p;
        rsum[r] += p;
      }
#pragma unroll
    for (int off = 1; off < 16; off <<= 1)
#pragma unroll
      for (int r = 0; r < 8; ++r) rsum[r] += __shfl_xor(rsum[r], off, 32);
#pragma unroll
    for (int r = 0; r < 8; ++r) lrow[r] = lrow[r] * corr[r] + rsum[r];
#pragma unroll
    for (int nt = 0; nt < 8; ++nt)
#pragma unroll
      for (int r = 0; r < 8; ++r) oacc[nt][r] *= corr[r];

    // ---- P (accumulator layout) -> LDS bf16 (A-fragment layout source)
#pragma unroll
    for (int nt = 0; nt < 4; ++nt)
#pragma unroll
      for (int r = 0; r < 8; ++r)
        Ps[w * 1024 + (lhalf * 8 + r) * 64 + nt * 16 + lrid] = f2bf(s[nt][r]);
    __syncthreads();  // cross-lane P visibility before fragment reload

    // ---- O += P * V : 8 n-tiles over d=128, 2 contraction steps over keys
#pragma unroll
    for (int kc = 0; kc < 2; ++kc) {
      v16bf ap = load_frag(&Ps[w * 1024 + lrid * 64 + kc * 32 + khalf]);
#pragma unroll
      for (int nt = 0; nt < 8; ++nt) {
        v16bf bv = load_frag(&Vts[(nt * 16 + lrid) * 64 + kc * 32 + khalf]);
        oacc[nt] = __builtin_amdgcn_wmma_f32_16x16x32_bf16(
            false, ap, false, bv, (short)0, oacc[nt], false, false);
      }
    }
  }

  // ---- normalize + store bf16 attention output
  bf16_t* Og = O + qrow0 * DIMC + h * HEADD;
#pragma unroll
  for (int r = 0; r < 8; ++r) {
    float inv = 1.0f / lrow[r];
    int m = w * 16 + lhalf * 8 + r;
#pragma unroll
    for (int nt = 0; nt < 8; ++nt)
      Og[(size_t)m * DIMC + nt * 16 + lrid] = f2bf(oacc[nt][r] * inv);
  }
}

// ---------------------------------------------------------------------------
// Host launcher
// ---------------------------------------------------------------------------
extern "C" void kernel_launch(void* const* d_in, const int* in_sizes, int n_in,
                              void* d_out, int out_size, void* d_ws,
                              size_t ws_size, hipStream_t stream) {
  (void)in_sizes; (void)n_in; (void)out_size; (void)ws_size;
  const float* enc = (const float*)d_in[0];
  const float* dec = (const float*)d_in[1];
  const float* Wkv = (const float*)d_in[2];
  const float* bkv = (const float*)d_in[3];
  const float* Wq  = (const float*)d_in[4];
  const float* bq  = (const float*)d_in[5];
  const float* Wo  = (const float*)d_in[6];
  const float* bo  = (const float*)d_in[7];
  float* out = (float*)d_out;

  char* ws = (char*)d_ws;
  bf16_t* Xenc = (bf16_t*)ws; ws += (size_t)MTOT * DIMC * sizeof(bf16_t);
  bf16_t* Xdec = (bf16_t*)ws; ws += (size_t)MTOT * DIMC * sizeof(bf16_t);
  bf16_t* Wkvt = (bf16_t*)ws; ws += (size_t)(2 * DIMC) * DIMC * sizeof(bf16_t);
  bf16_t* Wqt  = (bf16_t*)ws; ws += (size_t)DIMC * DIMC * sizeof(bf16_t);
  bf16_t* Wot  = (bf16_t*)ws; ws += (size_t)DIMC * DIMC * sizeof(bf16_t);
  bf16_t* KVb  = (bf16_t*)ws; ws += (size_t)MTOT * 2 * DIMC * sizeof(bf16_t);
  bf16_t* Qb   = (bf16_t*)ws; ws += (size_t)MTOT * DIMC * sizeof(bf16_t);
  bf16_t* AOb  = (bf16_t*)ws; ws += (size_t)MTOT * DIMC * sizeof(bf16_t);

  // 1) precision conversion (memory-bound, ~100 MB total)
  int nAct = MTOT * DIMC;
  cvt_bf16_kernel<<<nAct / 256, 256, 0, stream>>>(enc, Xenc, nAct);
  cvt_bf16_kernel<<<nAct / 256, 256, 0, stream>>>(dec, Xdec, nAct);
  cvt_transpose_kernel<<<(DIMC * 2 * DIMC) / 256, 256, 0, stream>>>(
      Wkv, Wkvt, DIMC, 2 * DIMC);
  cvt_transpose_kernel<<<(DIMC * DIMC) / 256, 256, 0, stream>>>(
      Wq, Wqt, DIMC, DIMC);
  cvt_transpose_kernel<<<(DIMC * DIMC) / 256, 256, 0, stream>>>(
      Wo, Wot, DIMC, DIMC);

  // 2) projections: KV = Xenc*Wkv+b (N=4096), Q = Xdec*Wq+b (N=2048)
  gemm_bf16_wmma_kernel<<<dim3((2 * DIMC) / 128, MTOT / 128), 256, 0, stream>>>(
      Xenc, Wkvt, bkv, nullptr, KVb, MTOT, 2 * DIMC, DIMC, 2 * DIMC);
  gemm_bf16_wmma_kernel<<<dim3(DIMC / 128, MTOT / 128), 256, 0, stream>>>(
      Xdec, Wqt, bq, nullptr, Qb, MTOT, DIMC, DIMC, DIMC);

  // 3) causal flash attention per (q-block, head, batch)
  flash_attn_wmma_kernel<<<dim3(SEQ / 64, NHEADS, BATCH), 128, 0, stream>>>(
      KVb, Qb, AOb);

  // 4) output projection (f32 result)
  gemm_bf16_wmma_kernel<<<dim3(DIMC / 128, MTOT / 128), 256, 0, stream>>>(
      AOb, Wot, bo, out, nullptr, MTOT, DIMC, DIMC, DIMC);
}
